// LearntPrototypes_42339787604062
// MI455X (gfx1250) — matrix operand order repaired and checked by reading, not compiled, not run on hardware
//
#include <hip/hip_runtime.h>

// Problem constants (from reference): B=16, C=256, H=W=128, K=256
#define C_DIM 256
#define K_DIM 256
#define HW_N  16384   // 128*128
#define HW_SH 14      // log2(HW)

typedef __attribute__((ext_vector_type(16))) __bf16  v16bf;
typedef __attribute__((ext_vector_type(8)))  float   v8f;

// Split fp32 into bf16 hi + bf16 lo using native converts (v_cvt_*_bf16_f32).
__device__ __forceinline__ void bf16_split(float v, __bf16& h, __bf16& l) {
    h = (__bf16)v;                 // RNE convert
    l = (__bf16)(v - (float)h);    // residual, converted RNE
}

// ---------------- prep kernel 1: p_sq[k] = sum_c protos[k][c]^2 ----------------
__global__ void lp_psq_kernel(const float* __restrict__ protos, float* __restrict__ psq) {
    int t = threadIdx.x;  // 256 threads, 1 block
    const float4* row = (const float4*)(protos + t * C_DIM);
    float s = 0.f;
    #pragma unroll 8
    for (int i = 0; i < C_DIM / 4; ++i) {
        float4 q = row[i];
        s += q.x * q.x + q.y * q.y + q.z * q.z + q.w * q.w;
    }
    psq[t] = s;
}

// ------- prep kernel 2: split protos to bf16 hi/lo in WMMA A-fragment order -------
// A-matrix (bf16, 16x32) per-lane layout (ISA 7.12.2):
//   lane m (0-15):  row m, slots 0..7 -> K=0..7,   slots 8..15 -> K=16..23
//   lane m+16:      row m, slots 0..7 -> K=8..15,  slots 8..15 -> K=24..31
// ws layout: pA[((pt*8 + chunk)*2 + plane)*32 + lane] : v16bf (32B per lane)
__global__ void lp_protoswz_kernel(const float* __restrict__ protos, v16bf* __restrict__ pA) {
    const int pt = blockIdx.x;       // 16 proto tiles
    const int t  = threadIdx.x;      // 256 threads
    const int chunk = t >> 5, l = t & 31, m = l & 15, half = l >> 4;
    v16bf hi, lo;
    #pragma unroll
    for (int slot = 0; slot < 16; ++slot) {
        int k = (slot & 7) + (half << 3) + ((slot >> 3) << 4);
        int c = (chunk << 5) + k;
        float v = protos[(pt * 16 + m) * C_DIM + c];
        __bf16 h, lw;
        bf16_split(v, h, lw);
        hi[slot] = h; lo[slot] = lw;
    }
    const int base = ((pt * 8 + chunk) * 2) * 32 + l;
    pA[base]      = hi;
    pA[base + 32] = lo;
}

// ---------------- main kernel: distances via bf16x3 WMMA GEMM ----------------
// Block: 256 threads (8 waves), tile = 32 spatial positions x all 256 prototypes.
// B-matrix (bf16, 32x16) per-lane layout:
//   lane n (0-15):  col n, slots 0..7 -> K=0..7,   slots 8..15 -> K=16..23
//   lane n+16:      col n, slots 0..7 -> K=8..15,  slots 8..15 -> K=24..31
__global__ __launch_bounds__(256)
void lp_main_kernel(const float* __restrict__ emb, const float* __restrict__ psq,
                    const v16bf* __restrict__ pA, float* __restrict__ out) {
    __shared__ v16bf ldsB[2 * 8 * 2 * 32];  // [plane][chunk][postile][lane] = 32 KB
    __shared__ float eSq[32];
    __shared__ float lpsq[K_DIM];

    const int t  = threadIdx.x;
    const int s0 = blockIdx.x * 32;          // global spatial index of tile
    const int b  = s0 >> HW_SH;
    const int sp = s0 & (HW_N - 1);
    const float* ebase = emb + (size_t)b * C_DIM * HW_N + sp;

    if (t < 32) eSq[t] = 0.f;
    lpsq[t] = psq[t];
    __syncthreads();

    // ---- stage embedding tile: fp32 -> bf16 hi/lo of (2*e), pre-swizzled B frags ----
    {
        const int chunk = t >> 5, l = t & 31, n0 = l & 15, half = l >> 4;
        const int cbase = (chunk << 5) + (half << 3);
        #pragma unroll
        for (int ptile = 0; ptile < 2; ++ptile) {
            const int n = n0 + ptile * 16;
            float part = 0.f;
            v16bf hi, lo;
            #pragma unroll
            for (int slot = 0; slot < 16; ++slot) {
                float v = ebase[(size_t)(cbase + (slot & 7) + ((slot >> 3) << 4)) * HW_N + n];   // coalesced 64B/half-wave
                part += v * v;
                __bf16 h, lw;
                bf16_split(v + v, h, lw);                             // fold the "2" into B
                hi[slot] = h; lo[slot] = lw;
            }
            ldsB[(chunk * 2 + ptile) * 32 + l]       = hi;  // plane 0 (2x ds_store_b128)
            ldsB[512 + (chunk * 2 + ptile) * 32 + l] = lo;  // plane 1
            atomicAdd(&eSq[n], part);                       // ds_add_f32
        }
    }
    __syncthreads();

    // ---- per-wave GEMM: 2 proto tiles x 2 position tiles ----
    const int wave = t >> 5;
    const int lane = t & 31;
    const int half = lane >> 4;
    const int n0   = lane & 15;
    const int pt0  = wave, pt1 = wave + 8;

    const float es0 = eSq[n0], es1 = eSq[n0 + 16];
    v8f acc00, acc01, acc10, acc11;
    #pragma unroll
    for (int i = 0; i < 8; ++i) {
        // D layout: VGPR i, lanes 0-15 -> M=i, lanes 16-31 -> M=i+8; N = lane%16
        float ps0 = lpsq[pt0 * 16 + (half << 3) + i];
        float ps1 = lpsq[pt1 * 16 + (half << 3) + i];
        acc00[i] = -(ps0 + es0);
        acc01[i] = -(ps0 + es1);
        acc10[i] = -(ps1 + es0);
        acc11[i] = -(ps1 + es1);
    }

    #pragma unroll
    for (int chunk = 0; chunk < 8; ++chunk) {
        v16bf bh0 = ldsB[(chunk * 2 + 0) * 32 + lane];
        v16bf bl0 = ldsB[512 + (chunk * 2 + 0) * 32 + lane];
        v16bf bh1 = ldsB[(chunk * 2 + 1) * 32 + lane];
        v16bf bl1 = ldsB[512 + (chunk * 2 + 1) * 32 + lane];

        const v16bf* a0p = pA + ((size_t)(pt0 * 8 + chunk) * 2) * 32 + lane;
        const v16bf* a1p = pA + ((size_t)(pt1 * 8 + chunk) * 2) * 32 + lane;
        v16bf a0h = a0p[0];
        v16bf a0l = a0p[32];
        v16bf a1h = a1p[0];
        v16bf a1l = a1p[32];
        if (chunk < 7) {                       // warm L0 for next chunk's A frags
            __builtin_prefetch(a0p + 64, 0, 3);
            __builtin_prefetch(a1p + 64, 0, 3);
        }

        // bf16x3: hi*hi + hi*lo + lo*hi  (error ~2^-16, fp32 accumulate)
        acc00 = __builtin_amdgcn_wmma_f32_16x16x32_bf16(false, a0h, false, bh0, (short)0, acc00, false, false);
        acc00 = __builtin_amdgcn_wmma_f32_16x16x32_bf16(false, a0h, false, bl0, (short)0, acc00, false, false);
        acc00 = __builtin_amdgcn_wmma_f32_16x16x32_bf16(false, a0l, false, bh0, (short)0, acc00, false, false);

        acc01 = __builtin_amdgcn_wmma_f32_16x16x32_bf16(false, a0h, false, bh1, (short)0, acc01, false, false);
        acc01 = __builtin_amdgcn_wmma_f32_16x16x32_bf16(false, a0h, false, bl1, (short)0, acc01, false, false);
        acc01 = __builtin_amdgcn_wmma_f32_16x16x32_bf16(false, a0l, false, bh1, (short)0, acc01, false, false);

        acc10 = __builtin_amdgcn_wmma_f32_16x16x32_bf16(false, a1h, false, bh0, (short)0, acc10, false, false);
        acc10 = __builtin_amdgcn_wmma_f32_16x16x32_bf16(false, a1h, false, bl0, (short)0, acc10, false, false);
        acc10 = __builtin_amdgcn_wmma_f32_16x16x32_bf16(false, a1l, false, bh0, (short)0, acc10, false, false);

        acc11 = __builtin_amdgcn_wmma_f32_16x16x32_bf16(false, a1h, false, bh1, (short)0, acc11, false, false);
        acc11 = __builtin_amdgcn_wmma_f32_16x16x32_bf16(false, a1h, false, bl1, (short)0, acc11, false, false);
        acc11 = __builtin_amdgcn_wmma_f32_16x16x32_bf16(false, a1l, false, bh1, (short)0, acc11, false, false);
    }

    // ---- epilogue: out = -max(dist,0) = min(acc, 0); 64B-contiguous per lane-half ----
    const size_t obase = (size_t)b * K_DIM * HW_N + sp;
    #pragma unroll
    for (int i = 0; i < 8; ++i) {
        const int p0 = pt0 * 16 + (half << 3) + i;
        const int p1 = pt1 * 16 + (half << 3) + i;
        out[obase + (size_t)p0 * HW_N + n0]      = fminf(acc00[i], 0.f);
        out[obase + (size_t)p0 * HW_N + 16 + n0] = fminf(acc01[i], 0.f);
        out[obase + (size_t)p1 * HW_N + n0]      = fminf(acc10[i], 0.f);
        out[obase + (size_t)p1 * HW_N + 16 + n0] = fminf(acc11[i], 0.f);
    }
}

extern "C" void kernel_launch(void* const* d_in, const int* in_sizes, int n_in,
                              void* d_out, int out_size, void* d_ws, size_t ws_size,
                              hipStream_t stream) {
    (void)in_sizes; (void)n_in; (void)out_size; (void)ws_size;
    const float* emb    = (const float*)d_in[0];   // [16,256,128,128] f32
    const float* protos = (const float*)d_in[1];   // [256,256] f32
    float* out = (float*)d_out;                    // [16,256,128,128] f32

    float* psq = (float*)d_ws;                          // 256 f32 = 1 KB
    v16bf* pA  = (v16bf*)((char*)d_ws + 1024);          // 16*8*2*32 v16bf = 256 KB

    lp_psq_kernel<<<1, 256, 0, stream>>>(protos, psq);
    lp_protoswz_kernel<<<16, 256, 0, stream>>>(protos, pA);
    lp_main_kernel<<<8192, 256, 0, stream>>>(emb, psq, pA, out);
}
